// AttentionPairBiasPairformerDeepspeed_2370821947559
// MI455X (gfx1250) — compile-verified, matrix operand order checked by reading, not run on hardware
//
#include <hip/hip_runtime.h>
#include <math.h>

typedef __attribute__((ext_vector_type(16))) __bf16 v16bf;
typedef __attribute__((ext_vector_type(8)))  float  v8f;
typedef __attribute__((ext_vector_type(2)))  float  v2f;

#define LSEQ 768
#define CA   384
#define CP   128
#define NH   16
#define CH   24
#define CHP  32   // head dim padded to WMMA K

union BV { uint4 u[2]; v16bf v; };

// ---------------------------------------------------------------------------
// LayerNorm(A) -> bf16, one wave per row
// ---------------------------------------------------------------------------
__global__ __launch_bounds__(32) void ln_a_kernel(const float* __restrict__ A,
                                                  const float* __restrict__ g,
                                                  const float* __restrict__ b,
                                                  __bf16* __restrict__ out) {
  int i = blockIdx.x;
  int lane = threadIdx.x;
  const float4* row = reinterpret_cast<const float4*>(A + (size_t)i * CA);
  float xs[12];
#pragma unroll
  for (int e = 0; e < 3; ++e) {
    float4 t = row[e * 32 + lane];
    xs[e * 4 + 0] = t.x; xs[e * 4 + 1] = t.y; xs[e * 4 + 2] = t.z; xs[e * 4 + 3] = t.w;
  }
  float s = 0.f, q = 0.f;
#pragma unroll
  for (int e = 0; e < 12; ++e) { s += xs[e]; q += xs[e] * xs[e]; }
#pragma unroll
  for (int m = 1; m < 32; m <<= 1) { s += __shfl_xor(s, m, 32); q += __shfl_xor(q, m, 32); }
  float mu = s * (1.0f / CA);
  float var = q * (1.0f / CA) - mu * mu;
  float rs = rsqrtf(var + 1e-5f);
#pragma unroll
  for (int e = 0; e < 3; ++e) {
#pragma unroll
    for (int c = 0; c < 4; ++c) {
      int col = e * 128 + lane * 4 + c;
      float v = (xs[e * 4 + c] - mu) * rs * g[col] + b[col];
      out[(size_t)i * CA + col] = (__bf16)v;
    }
  }
}

// ---------------------------------------------------------------------------
// Zero scratch (q/k pads, vt pads)
// ---------------------------------------------------------------------------
__global__ void zero_kernel(unsigned int* __restrict__ p, size_t n) {
  size_t i = (size_t)blockIdx.x * 256 + threadIdx.x;
  if (i < n) p[i] = 0u;
}

// ---------------------------------------------------------------------------
// Swizzle f32 [384,384] weight into WMMA bf16 B-operand lane layout
// tile (kt,nt): lane l, elem e -> W[kt*32 + Kmap(e,l/16)][nt*16 + l%16]
// ---------------------------------------------------------------------------
__global__ void swz_wbf_kernel(const float* __restrict__ W, __bf16* __restrict__ dst) {
  int tid = blockIdx.x * 256 + threadIdx.x;           // < 384*384
  int e = tid & 15;
  int lane = (tid >> 4) & 31;
  int nt = (tid >> 9) % 24;
  int kt = tid / (512 * 24);
  int half = lane >> 4;
  int K = kt * 32 + ((e < 8) ? (half * 8 + e) : (8 + half * 8 + e));
  int n = nt * 16 + (lane & 15);
  dst[tid] = (__bf16)W[(size_t)K * CA + n];
}

// Swizzle Wa (f32) into 16x16x4 f32 WMMA B layout: lane l, e -> K = kt*4 + 2*(l/16)+e
__global__ void swz_wa_kernel(const float* __restrict__ W, float* __restrict__ dst) {
  int tid = blockIdx.x * 256 + threadIdx.x;           // < 96*24*32*2
  int e = tid & 1;
  int lane = (tid >> 1) & 31;
  int nt = (tid >> 6) % 24;
  int kt = tid / 1536;
  int K = kt * 4 + (lane >> 4) * 2 + e;
  int n = nt * 16 + (lane & 15);
  dst[tid] = W[(size_t)K * CA + n];
}

// ---------------------------------------------------------------------------
// Projection GEMM: [768,384]bf16 x [384,384]bf16 -> epilogue per mode
//   mode 0/1: store bf16 padded [i][h][32] (q scaled, k unscaled)
//   mode 2  : store bf16 transposed Vt [h][32][768]
//   mode 3  : sigmoid -> f32 [i][384]
// one wave per 16x16 tile
// ---------------------------------------------------------------------------
__global__ __launch_bounds__(32) void proj_kernel(const __bf16* __restrict__ A,
                                                  const __bf16* __restrict__ Wsw,
                                                  __bf16* __restrict__ outb,
                                                  float* __restrict__ outf,
                                                  int mode, float scale) {
  int nt = blockIdx.x, it = blockIdx.y;
  int lane = threadIdx.x, half = lane >> 4, lm = lane & 15;
  int i0 = it * 16;
  v8f acc = {};
  const __bf16* arow = A + (size_t)(i0 + lm) * CA;
  for (int kt = 0; kt < 12; ++kt) {
    BV a, b;
    a.u[0] = *reinterpret_cast<const uint4*>(arow + kt * 32 + half * 8);
    a.u[1] = *reinterpret_cast<const uint4*>(arow + kt * 32 + 16 + half * 8);
    const uint4* bp = reinterpret_cast<const uint4*>(Wsw + ((size_t)(kt * 24 + nt) * 32 + lane) * 16);
    b.u[0] = bp[0]; b.u[1] = bp[1];
    acc = __builtin_amdgcn_wmma_f32_16x16x32_bf16(false, a.v, false, b.v, (short)0, acc, false, false);
  }
#pragma unroll
  for (int r = 0; r < 8; ++r) {
    int i = i0 + r + 8 * half;
    int n = nt * 16 + lm;
    float val = acc[r];
    if (mode <= 1) {
      int h = n / 24, c = n % 24;
      outb[(size_t)i * (NH * CHP) + h * CHP + c] = (__bf16)(val * scale);
    } else if (mode == 2) {
      int h = n / 24, c = n % 24;
      outb[(size_t)h * (CHP * LSEQ) + (size_t)c * LSEQ + i] = (__bf16)val;
    } else {
      outf[(size_t)i * CA + n] = 1.0f / (1.0f + __expf(-val));
    }
  }
}

// ---------------------------------------------------------------------------
// Pair bias: per (i,j): LN(Z[i,j,:128]) . Wb[:,h] + Beta[i,j] -> bias[h][i][j]
// 256-thread blocks, one wave per (i,j); Wb + gamma/beta cached in LDS
// ---------------------------------------------------------------------------
__global__ __launch_bounds__(256) void bias_kernel(const float* __restrict__ Z,
                                                   const float* __restrict__ Beta,
                                                   const float* __restrict__ Wb,
                                                   const float* __restrict__ g0,
                                                   const float* __restrict__ b0,
                                                   float* __restrict__ bias) {
  __shared__ float sWb[CP * NH];
  __shared__ float sg[CP];
  __shared__ float sb[CP];
  for (int t = threadIdx.x; t < CP * NH; t += 256) sWb[t] = Wb[t];
  if (threadIdx.x < CP) { sg[threadIdx.x] = g0[threadIdx.x]; sb[threadIdx.x] = b0[threadIdx.x]; }
  __syncthreads();

  int wid = threadIdx.x >> 5, lane = threadIdx.x & 31;
  size_t p = (size_t)blockIdx.x * 8 + wid;            // pair index i*768+j
  float4 z4 = reinterpret_cast<const float4*>(Z + p * CP)[lane];
  float zs[4] = {z4.x, z4.y, z4.z, z4.w};
  float s = zs[0] + zs[1] + zs[2] + zs[3];
  float q = zs[0]*zs[0] + zs[1]*zs[1] + zs[2]*zs[2] + zs[3]*zs[3];
#pragma unroll
  for (int m = 1; m < 32; m <<= 1) { s += __shfl_xor(s, m, 32); q += __shfl_xor(q, m, 32); }
  float mu = s * (1.0f / CP);
  float var = q * (1.0f / CP) - mu * mu;
  float rs = rsqrtf(var + 1e-5f);

  float acc[NH];
#pragma unroll
  for (int h = 0; h < NH; ++h) acc[h] = 0.f;
#pragma unroll
  for (int e = 0; e < 4; ++e) {
    int d = lane * 4 + e;
    float xn = (zs[e] - mu) * rs * sg[d] + sb[d];
    const float* wr = &sWb[d * NH];
#pragma unroll
    for (int h = 0; h < NH; ++h) acc[h] += xn * wr[h];
  }
#pragma unroll
  for (int h = 0; h < NH; ++h) {
#pragma unroll
    for (int m = 1; m < 32; m <<= 1) acc[h] += __shfl_xor(acc[h], m, 32);
  }
  float beta = Beta[p];
  if (lane < NH) bias[(size_t)lane * (LSEQ * LSEQ) + p] = acc[lane] + beta;
}

// ---------------------------------------------------------------------------
// Flash attention per (16-row i-tile, head). One wave per block.
// ---------------------------------------------------------------------------
__global__ __launch_bounds__(32) void attn_kernel(const __bf16* __restrict__ qp,
                                                  const __bf16* __restrict__ kp,
                                                  const __bf16* __restrict__ vt,
                                                  const float* __restrict__ bias,
                                                  const float* __restrict__ g,
                                                  float* __restrict__ obuf) {
  int i0 = blockIdx.x * 16;
  int h = blockIdx.y;
  int lane = threadIdx.x, half = lane >> 4, lm = lane & 15;
  __shared__ __bf16 sp[16 * 32];

  BV qa;
  const __bf16* qb = qp + (size_t)(i0 + lm) * (NH * CHP) + h * CHP;
  qa.u[0] = *reinterpret_cast<const uint4*>(qb + half * 8);
  qa.u[1] = *reinterpret_cast<const uint4*>(qb + 16 + half * 8);

  float rm[8], rl[8];
  v8f o0 = {}, o1 = {};
#pragma unroll
  for (int r = 0; r < 8; ++r) { rm[r] = -3.0e38f; rl[r] = 0.f; }
  const float* bh = bias + (size_t)h * LSEQ * LSEQ;

  for (int j0 = 0; j0 < LSEQ; j0 += 32) {
    BV kb0, kb1;
    const __bf16* k0 = kp + (size_t)(j0 + lm) * (NH * CHP) + h * CHP;
    const __bf16* k1 = kp + (size_t)(j0 + 16 + lm) * (NH * CHP) + h * CHP;
    kb0.u[0] = *reinterpret_cast<const uint4*>(k0 + half * 8);
    kb0.u[1] = *reinterpret_cast<const uint4*>(k0 + 16 + half * 8);
    kb1.u[0] = *reinterpret_cast<const uint4*>(k1 + half * 8);
    kb1.u[1] = *reinterpret_cast<const uint4*>(k1 + 16 + half * 8);

    v8f z = {};
    v8f s0 = __builtin_amdgcn_wmma_f32_16x16x32_bf16(false, qa.v, false, kb0.v, (short)0, z, false, false);
    v8f s1 = __builtin_amdgcn_wmma_f32_16x16x32_bf16(false, qa.v, false, kb1.v, (short)0, z, false, false);

#pragma unroll
    for (int r = 0; r < 8; ++r) {
      int i = i0 + r + 8 * half;
      s0[r] += bh[(size_t)i * LSEQ + j0 + lm];
      s1[r] += bh[(size_t)i * LSEQ + j0 + 16 + lm];
    }
    // online softmax: row reductions within each 16-lane half
#pragma unroll
    for (int r = 0; r < 8; ++r) {
      float t = fmaxf(s0[r], s1[r]);
      t = fmaxf(t, __shfl_xor(t, 1, 32));
      t = fmaxf(t, __shfl_xor(t, 2, 32));
      t = fmaxf(t, __shfl_xor(t, 4, 32));
      t = fmaxf(t, __shfl_xor(t, 8, 32));
      float mn = fmaxf(rm[r], t);
      float corr = __expf(rm[r] - mn);
      rm[r] = mn;
      float p0 = __expf(s0[r] - mn);
      float p1 = __expf(s1[r] - mn);
      float rsum = p0 + p1;
      rsum += __shfl_xor(rsum, 1, 32);
      rsum += __shfl_xor(rsum, 2, 32);
      rsum += __shfl_xor(rsum, 4, 32);
      rsum += __shfl_xor(rsum, 8, 32);
      rl[r] = rl[r] * corr + rsum;
      o0[r] *= corr; o1[r] *= corr;
      int m = r + 8 * half;
      sp[m * 32 + lm] = (__bf16)p0;
      sp[m * 32 + 16 + lm] = (__bf16)p1;
    }
    // re-shape P: C-layout (via LDS) -> A-layout (DS in-order within a wave)
    BV pa;
    pa.u[0] = *reinterpret_cast<const uint4*>(&sp[lm * 32 + half * 8]);
    pa.u[1] = *reinterpret_cast<const uint4*>(&sp[lm * 32 + 16 + half * 8]);

    BV vb0, vb1;
    const __bf16* v0 = vt + (size_t)h * (CHP * LSEQ) + (size_t)lm * LSEQ + j0;
    const __bf16* v1 = vt + (size_t)h * (CHP * LSEQ) + (size_t)(16 + lm) * LSEQ + j0;
    vb0.u[0] = *reinterpret_cast<const uint4*>(v0 + half * 8);
    vb0.u[1] = *reinterpret_cast<const uint4*>(v0 + 16 + half * 8);
    vb1.u[0] = *reinterpret_cast<const uint4*>(v1 + half * 8);
    vb1.u[1] = *reinterpret_cast<const uint4*>(v1 + 16 + half * 8);

    o0 = __builtin_amdgcn_wmma_f32_16x16x32_bf16(false, pa.v, false, vb0.v, (short)0, o0, false, false);
    o1 = __builtin_amdgcn_wmma_f32_16x16x32_bf16(false, pa.v, false, vb1.v, (short)0, o1, false, false);
  }
#pragma unroll
  for (int r = 0; r < 8; ++r) {
    int i = i0 + r + 8 * half;
    float inv = 1.0f / rl[r];
    int n = h * CH + lm;
    obuf[(size_t)i * CA + n] = o0[r] * inv * g[(size_t)i * CA + n];
    if (lm < 8) {
      int n2 = h * CH + 16 + lm;
      obuf[(size_t)i * CA + n2] = o1[r] * inv * g[(size_t)i * CA + n2];
    }
  }
}

// ---------------------------------------------------------------------------
// Output projection in f32 via V_WMMA_F32_16X16X4_F32
// ---------------------------------------------------------------------------
__global__ __launch_bounds__(32) void out_gemm_kernel(const float* __restrict__ O,
                                                      const float* __restrict__ Wsw,
                                                      float* __restrict__ out) {
  int nt = blockIdx.x, it = blockIdx.y;
  int lane = threadIdx.x, half = lane >> 4, lm = lane & 15;
  int i0 = it * 16;
  v8f acc = {};
  const float* orow = O + (size_t)(i0 + lm) * CA;
  for (int kt = 0; kt < 96; ++kt) {
    v2f a = *reinterpret_cast<const v2f*>(orow + kt * 4 + half * 2);
    v2f b = *reinterpret_cast<const v2f*>(Wsw + ((size_t)(kt * 24 + nt) * 32 + lane) * 2);
    acc = __builtin_amdgcn_wmma_f32_16x16x4_f32(false, a, false, b, (short)0, acc, false, false);
  }
#pragma unroll
  for (int r = 0; r < 8; ++r)
    out[(size_t)(i0 + r + 8 * half) * CA + nt * 16 + lm] = acc[r];
}

// ---------------------------------------------------------------------------
extern "C" void kernel_launch(void* const* d_in, const int* in_sizes, int n_in,
                              void* d_out, int out_size, void* d_ws, size_t ws_size,
                              hipStream_t stream) {
  const float* A_I  = (const float*)d_in[0];
  const float* Z_II = (const float*)d_in[1];
  const float* Beta = (const float*)d_in[2];
  const float* Wq   = (const float*)d_in[3];
  const float* Wk   = (const float*)d_in[4];
  const float* Wv   = (const float*)d_in[5];
  const float* Wg   = (const float*)d_in[6];
  const float* Wb   = (const float*)d_in[7];
  const float* Wa   = (const float*)d_in[8];
  const float* ln0g = (const float*)d_in[9];
  const float* ln0b = (const float*)d_in[10];
  const float* ln1g = (const float*)d_in[11];
  const float* ln1b = (const float*)d_in[12];
  float* out = (float*)d_out;

  char* w = (char*)d_ws;
  size_t off = 0;
  auto alloc = [&](size_t bytes) -> char* {
    char* p = w + off;
    off = (off + bytes + 255) & ~(size_t)255;
    return p;
  };
  __bf16* a_bf = (__bf16*)alloc((size_t)LSEQ * CA * 2);
  __bf16* qp   = (__bf16*)alloc((size_t)LSEQ * NH * CHP * 2);
  __bf16* kp   = (__bf16*)alloc((size_t)LSEQ * NH * CHP * 2);
  __bf16* vt   = (__bf16*)alloc((size_t)NH * CHP * LSEQ * 2);
  float*  gbuf = (float*)alloc((size_t)LSEQ * CA * 4);
  __bf16* wqs  = (__bf16*)alloc((size_t)CA * CA * 2);
  __bf16* wks  = (__bf16*)alloc((size_t)CA * CA * 2);
  __bf16* wvs  = (__bf16*)alloc((size_t)CA * CA * 2);
  __bf16* wgs  = (__bf16*)alloc((size_t)CA * CA * 2);
  float*  was  = (float*)alloc((size_t)CA * CA * 4);
  float*  bias = (float*)alloc((size_t)NH * LSEQ * LSEQ * 4);
  float*  obuf = (float*)alloc((size_t)LSEQ * CA * 4);
  (void)in_sizes; (void)n_in; (void)out_size; (void)ws_size;

  ln_a_kernel<<<LSEQ, 32, 0, stream>>>(A_I, ln1g, ln1b, a_bf);

  size_t zn = ((size_t)LSEQ * NH * CHP * 2 * 2 + (size_t)NH * CHP * LSEQ * 2) / 4; // qp+kp+vt in uints
  zero_kernel<<<(unsigned)((zn + 255) / 256), 256, 0, stream>>>((unsigned int*)qp, zn);

  swz_wbf_kernel<<<576, 256, 0, stream>>>(Wq, wqs);
  swz_wbf_kernel<<<576, 256, 0, stream>>>(Wk, wks);
  swz_wbf_kernel<<<576, 256, 0, stream>>>(Wv, wvs);
  swz_wbf_kernel<<<576, 256, 0, stream>>>(Wg, wgs);
  swz_wa_kernel<<<576, 256, 0, stream>>>(Wa, was);

  const float scale = 0.2041241452319315f; // 1/sqrt(24)
  dim3 pg(24, 48);
  proj_kernel<<<pg, 32, 0, stream>>>(a_bf, wqs, qp, nullptr, 0, scale);
  proj_kernel<<<pg, 32, 0, stream>>>(a_bf, wks, kp, nullptr, 1, 1.0f);
  proj_kernel<<<pg, 32, 0, stream>>>(a_bf, wvs, vt, nullptr, 2, 1.0f);
  proj_kernel<<<pg, 32, 0, stream>>>(a_bf, wgs, nullptr, gbuf, 3, 1.0f);

  bias_kernel<<<(LSEQ * LSEQ) / 8, 256, 0, stream>>>(Z_II, Beta, Wb, ln0g, ln0b, bias);

  attn_kernel<<<dim3(48, 16), 32, 0, stream>>>(qp, kp, vt, bias, gbuf, obuf);

  out_gemm_kernel<<<pg, 32, 0, stream>>>(obuf, was, out);
}
